// HeteroHGT_33303176413373
// MI455X (gfx1250) — compile-verified, matrix operand order checked by reading, not run on hardware
//
#include <hip/hip_runtime.h>
#include <stdint.h>

// ---------------------------------------------------------------------------
// Types
// ---------------------------------------------------------------------------
typedef __attribute__((ext_vector_type(16))) __bf16 v16bf;
typedef __attribute__((ext_vector_type(8)))  __bf16 v8bf;
typedef __attribute__((ext_vector_type(8)))  float  v8f;
typedef __attribute__((ext_vector_type(4)))  unsigned int u32x4;
typedef __attribute__((ext_vector_type(8)))  int    i32x8;
typedef __attribute__((ext_vector_type(4)))  int    i32x4;

#if __has_builtin(__builtin_amdgcn_tensor_load_to_lds) && __has_builtin(__builtin_amdgcn_s_wait_tensorcnt)
#define USE_TDM 1
#else
#define USE_TDM 0
#endif

#define LDST 40   // LDS row stride in bf16 elems: 80B -> conflict-free frag reads, 16B aligned

// ---------------------------------------------------------------------------
// Elementwise helpers
// ---------------------------------------------------------------------------
__global__ void cast_bf16_kernel(const float* __restrict__ x, __bf16* __restrict__ y, long n) {
    long i = (long)blockIdx.x * blockDim.x + threadIdx.x;
    long stride = (long)gridDim.x * blockDim.x;
    for (; i < n; i += stride) y[i] = (__bf16)x[i];
}

// out[n*K + k] = (n < Nreal) ? in[k*Nreal + n] : 0   (weight -> Bt[Npad][K] bf16)
__global__ void transpose_cast_bf16_kernel(const float* __restrict__ in, __bf16* __restrict__ out,
                                           int K, int Nreal, int Npad) {
    long total = (long)Npad * K;
    long i = (long)blockIdx.x * blockDim.x + threadIdx.x;
    long stride = (long)gridDim.x * blockDim.x;
    for (; i < total; i += stride) {
        int n = (int)(i / K);
        int k = (int)(i % K);
        float v = (n < Nreal) ? in[(size_t)k * Nreal + n] : 0.f;
        out[i] = (__bf16)v;
    }
}

__global__ void gelu_bf16_kernel(const float* __restrict__ x, __bf16* __restrict__ y, long n) {
    long i = (long)blockIdx.x * blockDim.x + threadIdx.x;
    long stride = (long)gridDim.x * blockDim.x;
    for (; i < n; i += stride) {
        float v = x[i];
        y[i] = (__bf16)(0.5f * v * (1.0f + erff(v * 0.7071067811865476f)));
    }
}

__global__ void fill_f32_kernel(float* __restrict__ p, float v, long n) {
    long i = (long)blockIdx.x * blockDim.x + threadIdx.x;
    long stride = (long)gridDim.x * blockDim.x;
    for (; i < n; i += stride) p[i] = v;
}

__global__ void fill_u32_kernel(unsigned* __restrict__ p, unsigned v, long n) {
    long i = (long)blockIdx.x * blockDim.x + threadIdx.x;
    long stride = (long)gridDim.x * blockDim.x;
    for (; i < n; i += stride) p[i] = v;
}

// out[j] = sum_d b[d] * B[d*128 + j]   (folded bias:  b' = b @ B)
__global__ void vecmat128_kernel(const float* __restrict__ b, const float* __restrict__ B,
                                 float* __restrict__ out) {
    int j = threadIdx.x;
    float s = 0.f;
    for (int d = 0; d < 128; ++d) s += b[d] * B[d * 128 + j];
    out[j] = s;
}

// ---------------------------------------------------------------------------
// Tensor Data Mover: async 2D bf16 tile (tile_rows x 32) -> LDS, with LDS
// padding of 4 DWORDs every 16 DWORDs (64B row -> 80B LDS stride == LDST).
// ---------------------------------------------------------------------------
#if USE_TDM
__device__ __forceinline__ void tdm_load_tile_bf16(unsigned lds_off, const __bf16* gptr,
                                                   int rows_rem, int row_len_elems,
                                                   int tile_rows, int tile_cols,
                                                   long row_stride_elems) {
    unsigned long long ga = (unsigned long long)(uintptr_t)gptr;
    u32x4 g0;
    g0[0] = 1u;                                          // count=1, user descriptor
    g0[1] = lds_off;                                     // LDS byte address
    g0[2] = (unsigned)(ga & 0xFFFFFFFFull);              // global_addr[31:0]
    g0[3] = (unsigned)((ga >> 32) & 0x01FFFFFFull) | (2u << 30);  // addr[56:32] | type=2
    unsigned td0 = (unsigned)row_len_elems;
    unsigned td1 = (unsigned)(rows_rem > 0 ? rows_rem : 0);
    unsigned long long s0 = (unsigned long long)row_stride_elems;
    i32x8 g1;
    // data_size=1 (2B) | pad_enable | pad_interval=3 (16 DW) | pad_amount=3 (4 DW)
    g1[0] = (int)((1u << 16) | (1u << 20) | (3u << 22) | (3u << 25));
    g1[1] = (int)((td0 & 0xFFFFu) << 16);                // tensor_dim0[15:0]
    g1[2] = (int)(((td0 >> 16) & 0xFFFFu) | ((td1 & 0xFFFFu) << 16));
    g1[3] = (int)(((td1 >> 16) & 0xFFFFu) | (((unsigned)tile_cols & 0xFFFFu) << 16));
    g1[4] = (int)((unsigned)tile_rows & 0xFFFFu);        // tile_dim1; tile_dim2 = 0
    g1[5] = (int)(s0 & 0xFFFFFFFFull);                   // tensor_dim0_stride lo
    g1[6] = (int)((s0 >> 32) & 0xFFFFull);               // stride hi; dim1_stride = 0
    g1[7] = 0;
    i32x4 gz; gz[0] = 0; gz[1] = 0; gz[2] = 0; gz[3] = 0;
#if defined(__clang_major__) && (__clang_major__ >= 23)
    i32x8 gz8; for (int z = 0; z < 8; ++z) gz8[z] = 0;
    __builtin_amdgcn_tensor_load_to_lds(g0, g1, gz, gz, gz8, 0);
#else
    __builtin_amdgcn_tensor_load_to_lds(g0, g1, gz, gz, 0);
#endif
}
#endif

// ---------------------------------------------------------------------------
// WMMA fragment load: 16 rows x 32 K, bf16, LDS tile with stride LDST.
// Operand layout per CDNA5 ISA: lanes 0-15 -> index, lanes 16-31 -> K upper
// half; each lane holds K {kb..kb+7} then {kb+16..kb+23}, kb=(lane>>4)*8.
// ---------------------------------------------------------------------------
__device__ __forceinline__ v16bf load_frag(const __bf16* base, int idx, int lane) {
    int r = idx + (lane & 15);
    int kb = (lane >> 4) * 8;
    const __bf16* p = base + r * LDST + kb;
    v8bf lo = *(const v8bf*)(p);
    v8bf hi = *(const v8bf*)(p + 16);
    return __builtin_shufflevector(lo, hi, 0, 1, 2, 3, 4, 5, 6, 7,
                                           8, 9, 10, 11, 12, 13, 14, 15);
}

__device__ __forceinline__ v8f v8f_zero() {
    v8f z;
#pragma unroll
    for (int i = 0; i < 8; ++i) z[i] = 0.f;
    return z;
}

__device__ __forceinline__ v8bf v8bf_zero() {
    v8bf z;
#pragma unroll
    for (int i = 0; i < 8; ++i) z[i] = (__bf16)0.f;
    return z;
}

// Manual (fallback) staging of a bf16 tile (tile_rows x 32) into padded LDS
__device__ __forceinline__ void stage_tile_manual(__bf16* dst, const __bf16* src,
                                                  int tile_rows, int rows_rem,
                                                  size_t rstride, int tid) {
    for (int c = tid; c < tile_rows * 4; c += 256) {
        int rr = c >> 2;
        int ko = (c & 3) * 8;
        v8bf v = v8bf_zero();
        if (rr < rows_rem) v = *(const v8bf*)(src + (size_t)rr * rstride + ko);
        *(v8bf*)&dst[rr * LDST + ko] = v;
    }
}

// ---------------------------------------------------------------------------
// Generic WMMA GEMM:  C[M,N] = A_bf16[M,K] @ Bt_bf16[128,K]^T + bias
//   - block tile 128x128, 8 waves (4x2), wave tile 32x64 -> 8 WMMA / K-step
//   - A and B tiles streamed by the Tensor Data Mover, double buffered
//   - optional epilogue: sigmoid(gate) residual blend, f32 + bf16 outputs
// ---------------------------------------------------------------------------
#define BM 128
#define BN 128
#define BKK 32

__global__ __launch_bounds__(256) void gemm_bf16_wmma(
    const __bf16* __restrict__ A, const __bf16* __restrict__ Bt,
    const float* __restrict__ bias, const float* __restrict__ res,
    const float* __restrict__ gate,
    float* __restrict__ Cf, __bf16* __restrict__ Cbf,
    int M, int N, int Npad, int K)
{
    __shared__ __align__(16) __bf16 As[2][BM * LDST];
    __shared__ __align__(16) __bf16 Bs[2][BN * LDST];

    const int tid  = threadIdx.x;
    const int lane = tid & 31;
    const int wave = tid >> 5;
    const int wr = wave >> 1;   // 0..3
    const int wc = wave & 1;    // 0..1
    const int mblk = blockIdx.y * BM;
    const int nblk = blockIdx.x * BN;
    const int KT = K / BKK;

    v8f acc[2][4];
#pragma unroll
    for (int i = 0; i < 2; ++i)
#pragma unroll
        for (int j = 0; j < 4; ++j) acc[i][j] = v8f_zero();

#if USE_TDM
    if (wave == 0) {
        tdm_load_tile_bf16((unsigned)(uintptr_t)(&As[0][0]),
                           A + (size_t)mblk * K, M - mblk, K, BM, BKK, K);
        tdm_load_tile_bf16((unsigned)(uintptr_t)(&Bs[0][0]),
                           Bt + (size_t)nblk * K, Npad - nblk, K, BN, BKK, K);
    }
#endif

    for (int kt = 0; kt < KT; ++kt) {
        const int k0 = kt * BKK;
        const int buf = kt & 1;
        __syncthreads();   // previous compute done before buffers are rewritten

#if USE_TDM
        if (wave == 0) {
            if (kt + 1 < KT) {
                const int k1 = k0 + BKK;
                tdm_load_tile_bf16((unsigned)(uintptr_t)(&As[(kt + 1) & 1][0]),
                                   A + (size_t)mblk * K + k1, M - mblk, K - k1, BM, BKK, K);
                tdm_load_tile_bf16((unsigned)(uintptr_t)(&Bs[(kt + 1) & 1][0]),
                                   Bt + (size_t)nblk * K + k1, Npad - nblk, K - k1, BN, BKK, K);
                __builtin_amdgcn_s_wait_tensorcnt(2);   // current A+B resident
            } else {
                __builtin_amdgcn_s_wait_tensorcnt(0);
            }
        }
#else
        stage_tile_manual(&As[buf][0], A + (size_t)mblk * K + k0, BM, M - mblk, K, tid);
        stage_tile_manual(&Bs[buf][0], Bt + (size_t)nblk * K + k0, BN, Npad - nblk, K, tid);
#endif
        __syncthreads();   // LDS tiles visible to all waves

        // ---- WMMA compute: wave tile 32x64 = 2x4 of 16x16 ----
        v16bf af[2], bfm[4];
#pragma unroll
        for (int i = 0; i < 2; ++i) af[i]  = load_frag(&As[buf][0], wr * 32 + i * 16, lane);
#pragma unroll
        for (int j = 0; j < 4; ++j) bfm[j] = load_frag(&Bs[buf][0], wc * 64 + j * 16, lane);
#pragma unroll
        for (int i = 0; i < 2; ++i)
#pragma unroll
            for (int j = 0; j < 4; ++j)
                acc[i][j] = __builtin_amdgcn_wmma_f32_16x16x32_bf16(
                    false, af[i], false, bfm[j], (short)0, acc[i][j], false, false);
    }

    // ---- epilogue ----
    float g = 1.f, rw = 0.f;
    if (gate) { g = 1.f / (1.f + __expf(-gate[0])); rw = 1.f - g; }
    const int r = lane & 15;
    const int half = lane >> 4;
#pragma unroll
    for (int i = 0; i < 2; ++i) {
#pragma unroll
        for (int j = 0; j < 4; ++j) {
            int col = nblk + wc * 64 + j * 16 + r;
            if (col >= N) continue;
            float bv = bias ? bias[col] : 0.f;
#pragma unroll
            for (int v = 0; v < 8; ++v) {
                int row = mblk + wr * 32 + i * 16 + half * 8 + v;
                if (row >= M) continue;
                float val = acc[i][j][v] + bv;
                size_t o = (size_t)row * N + col;
                if (gate) val = g * val + rw * res[o];
                if (Cf)  Cf[o]  = val;
                if (Cbf) Cbf[o] = (__bf16)val;
            }
        }
    }
}

// ---------------------------------------------------------------------------
// Link-predictor GEMM (narrow N): C[M,5] = concat(GA0[gi0], GA1[gi1]) @ Bt^T
//   - block tile 256x16, 8 waves stacked on M, 2 WMMA per wave per K-step
//   - row indices cached in LDS; A gathered (bf16, 16B chunks); B via TDM
// ---------------------------------------------------------------------------
#define LBM 256

__global__ __launch_bounds__(256) void link_gemm_wmma(
    const __bf16* __restrict__ GA0, const __bf16* __restrict__ GA1,
    const int* __restrict__ gi0, const int* __restrict__ gi1,
    const __bf16* __restrict__ Bt,    // [16][K]
    const float* __restrict__ bias,
    float* __restrict__ Cf,
    int M, int N, int K)              // N = 5, K = 256
{
    __shared__ __align__(16) __bf16 As[2][LBM * LDST];
    __shared__ __align__(16) __bf16 Bs[2][16 * LDST];
    __shared__ int idx0[LBM];
    __shared__ int idx1[LBM];

    const int tid  = threadIdx.x;
    const int lane = tid & 31;
    const int wave = tid >> 5;
    const int mblk = blockIdx.y * LBM;
    const int KT = K / BKK;           // 8

    // cache the block's row indices in LDS (one global load per row)
    for (int rr = tid; rr < LBM; rr += 256) {
        int row = mblk + rr;
        idx0[rr] = (row < M) ? gi0[row] : 0;
        idx1[rr] = (row < M) ? gi1[row] : 0;
    }

    v8f acc[2];
#pragma unroll
    for (int i = 0; i < 2; ++i) acc[i] = v8f_zero();

#if USE_TDM
    if (wave == 0)
        tdm_load_tile_bf16((unsigned)(uintptr_t)(&Bs[0][0]), Bt, 16, K, 16, BKK, K);
#endif

    for (int kt = 0; kt < KT; ++kt) {
        const int k0 = kt * BKK;
        const int buf = kt & 1;
        __syncthreads();

        // ---- gather A tile: 256 rows x 32 K (1024 chunks of 8 bf16) ----
        for (int c = tid; c < LBM * 4; c += 256) {
            int rr = c >> 2;
            int ko = (c & 3) * 8;
            int row = mblk + rr;
            v8bf v = v8bf_zero();
            if (row < M) {
                const __bf16* srcp;
                if (k0 < 128) srcp = GA0 + (size_t)idx0[rr] * 128 + (k0 + ko);
                else          srcp = GA1 + (size_t)idx1[rr] * 128 + (k0 - 128 + ko);
                v = *(const v8bf*)srcp;
            }
            *(v8bf*)&As[buf][rr * LDST + ko] = v;
        }

#if USE_TDM
        if (wave == 0) {
            if (kt + 1 < KT) {
                tdm_load_tile_bf16((unsigned)(uintptr_t)(&Bs[(kt + 1) & 1][0]),
                                   Bt + (k0 + BKK), 16, K - (k0 + BKK), 16, BKK, K);
                __builtin_amdgcn_s_wait_tensorcnt(1);
            } else {
                __builtin_amdgcn_s_wait_tensorcnt(0);
            }
        }
#else
        stage_tile_manual(&Bs[buf][0], Bt + k0, 16, 16, K, tid);
#endif
        __syncthreads();

        // ---- WMMA: wave rows [wave*32, +32), cols [0,16) ----
        v16bf bfm = load_frag(&Bs[buf][0], 0, lane);
#pragma unroll
        for (int i = 0; i < 2; ++i) {
            v16bf af = load_frag(&As[buf][0], wave * 32 + i * 16, lane);
            acc[i] = __builtin_amdgcn_wmma_f32_16x16x32_bf16(
                false, af, false, bfm, (short)0, acc[i], false, false);
        }
    }

    // ---- epilogue ----
    const int r = lane & 15;
    const int half = lane >> 4;
    if (r < N) {
        float bv = bias ? bias[r] : 0.f;
#pragma unroll
        for (int i = 0; i < 2; ++i) {
#pragma unroll
            for (int v = 0; v < 8; ++v) {
                int row = mblk + wave * 32 + i * 16 + half * 8 + v;
                if (row >= M) continue;
                Cf[(size_t)row * N + r] = acc[i][v] + bv;
            }
        }
    }
}

// ---------------------------------------------------------------------------
// Edge phase (segment softmax + scatter), one wave per edge
// ---------------------------------------------------------------------------
#define NEG_INF_KEY 0x007FFFFFu   // monotonic key of -inf

__global__ void edge_alpha_max_kernel(const float* __restrict__ q, const float* __restrict__ k,
                                      const int* __restrict__ src, const int* __restrict__ dst,
                                      const float* __restrict__ pscale,
                                      float* __restrict__ alpha, unsigned* __restrict__ maxkey,
                                      int nE) {
    int e = (int)((blockIdx.x * (long)blockDim.x + threadIdx.x) >> 5);
    int lane = threadIdx.x & 31;
    if (e >= nE) return;
    int s = src[e], d = dst[e];
    float4 qv = *(const float4*)(q + (size_t)d * 128 + lane * 4);
    float4 kv = *(const float4*)(k + (size_t)s * 128 + lane * 4);
    float t = qv.x * kv.x + qv.y * kv.y + qv.z * kv.z + qv.w * kv.w;
#pragma unroll
    for (int off = 16; off > 0; off >>= 1) t += __shfl_xor(t, off, 32);
    float a = t * pscale[0] * 0.08838834764831845f;   // * p / sqrt(128)
    if (lane == 0) {
        alpha[e] = a;
        unsigned key = __float_as_uint(a);
        key = (key & 0x80000000u) ? ~key : (key | 0x80000000u);
        atomicMax(maxkey + d, key);
    }
}

__global__ void edge_exp_sum_kernel(const float* __restrict__ alpha, const int* __restrict__ dst,
                                    const unsigned* __restrict__ maxkey,
                                    float* __restrict__ ex, float* __restrict__ ssum, int nE) {
    int e = blockIdx.x * blockDim.x + threadIdx.x;
    if (e >= nE) return;
    int d = dst[e];
    unsigned key = maxkey[d];
    unsigned bits = (key & 0x80000000u) ? (key & 0x7FFFFFFFu) : ~key;
    float mx = __uint_as_float(bits);
    float v = __expf(alpha[e] - mx);
    ex[e] = v;
    atomicAdd(ssum + d, v);
}

__global__ void edge_scatter_kernel(const float* __restrict__ vp,
                                    const int* __restrict__ src, const int* __restrict__ dst,
                                    const float* __restrict__ ex, const float* __restrict__ ssum,
                                    float* __restrict__ agg, int nE) {
    int e = (int)((blockIdx.x * (long)blockDim.x + threadIdx.x) >> 5);
    int lane = threadIdx.x & 31;
    if (e >= nE) return;
    int s = src[e], d = dst[e];
    float w = ex[e] / ssum[d];
    float4 vv = *(const float4*)(vp + (size_t)s * 128 + lane * 4);
    float* o = agg + (size_t)d * 128 + lane * 4;
    atomicAdd(o + 0, vv.x * w);
    atomicAdd(o + 1, vv.y * w);
    atomicAdd(o + 2, vv.z * w);
    atomicAdd(o + 3, vv.w * w);
}

// ---------------------------------------------------------------------------
// Host side
// ---------------------------------------------------------------------------
extern "C" void kernel_launch(void* const* d_in, const int* in_sizes, int n_in,
                              void* d_out, int out_size, void* d_ws, size_t ws_size,
                              hipStream_t stream) {
    (void)in_sizes; (void)n_in; (void)out_size; (void)ws_size;

    const int ND = 20000, NPn = 20000, nE = 500000;
    const int FD = 1024, FP = 1280, HID = 128, EL = 200000, OUT = 5;

    const float* x_drug  = (const float*)d_in[0];
    const float* x_prot  = (const float*)d_in[1];
    const int*   e_dp    = (const int*)d_in[2];
    const int*   e_pd    = (const int*)d_in[3];
    const int*   e_lbl   = (const int*)d_in[4];
    const float* Wi_d    = (const float*)d_in[5];
    const float* bi_d    = (const float*)d_in[6];
    const float* Wi_p    = (const float*)d_in[7];
    const float* bi_p    = (const float*)d_in[8];
    const float* Wk      = (const float*)d_in[9];
    const float* bk      = (const float*)d_in[10];
    const float* Wq      = (const float*)d_in[11];
    const float* bq      = (const float*)d_in[12];
    const float* Wv      = (const float*)d_in[13];
    const float* bv      = (const float*)d_in[14];
    const float* a_rel   = (const float*)d_in[15];
    const float* m_rel   = (const float*)d_in[16];
    const float* p_rel   = (const float*)d_in[17];
    const float* Wo      = (const float*)d_in[18];
    const float* bo      = (const float*)d_in[19];
    const float* skip    = (const float*)d_in[20];
    const float* Wl      = (const float*)d_in[21];
    const float* bl      = (const float*)d_in[22];
    float* out = (float*)d_out;

    // ---- workspace bump allocator ----
    char* wbase = (char*)d_ws;
    size_t off = 0;
    auto alloc = [&](size_t bytes) -> void* {
        void* p = wbase + off;
        off = (off + bytes + 255) & ~(size_t)255;
        return p;
    };
    __bf16* xdrug_bf = (__bf16*)alloc((size_t)ND * FD * 2);
    __bf16* xprot_bf = (__bf16*)alloc((size_t)NPn * FP * 2);
    __bf16* Wk_bf    = (__bf16*)alloc(2 * 128 * 128 * 2);
    __bf16* Wv_bf    = (__bf16*)alloc(2 * 128 * 128 * 2);
    // transposed bf16 weights (Bt layout [Npad][K])
    __bf16* WiT_d    = (__bf16*)alloc((size_t)128 * FD * 2);
    __bf16* WiT_p    = (__bf16*)alloc((size_t)128 * FP * 2);
    __bf16* WqT      = (__bf16*)alloc(2 * 128 * 128 * 2);
    __bf16* WoT      = (__bf16*)alloc(2 * 128 * 128 * 2);
    __bf16* aT       = (__bf16*)alloc(2 * 128 * 128 * 2);
    __bf16* mT       = (__bf16*)alloc(2 * 128 * 128 * 2);
    __bf16* WkaT     = (__bf16*)alloc(2 * 128 * 128 * 2);
    __bf16* WvmT     = (__bf16*)alloc(2 * 128 * 128 * 2);
    __bf16* WlT      = (__bf16*)alloc((size_t)16 * 256 * 2);
    float*  Wka      = (float*)alloc(2 * 128 * 128 * 4);
    float*  Wvm      = (float*)alloc(2 * 128 * 128 * 4);
    float*  bka      = (float*)alloc(2 * 128 * 4);
    float*  bvm      = (float*)alloc(2 * 128 * 4);
    float*  xd_f     = (float*)alloc((size_t)ND * HID * 4);
    float*  xp_f     = (float*)alloc((size_t)NPn * HID * 4);
    __bf16* xd_bf    = (__bf16*)alloc((size_t)ND * HID * 2);
    __bf16* xp_bf    = (__bf16*)alloc((size_t)NPn * HID * 2);
    float*  q_d      = (float*)alloc((size_t)ND * HID * 4);
    float*  q_p      = (float*)alloc((size_t)NPn * HID * 4);
    float*  kp_d     = (float*)alloc((size_t)ND * HID * 4);
    float*  kp_p     = (float*)alloc((size_t)NPn * HID * 4);
    float*  vp_d     = (float*)alloc((size_t)ND * HID * 4);
    float*  vp_p     = (float*)alloc((size_t)NPn * HID * 4);
    float*  alpha    = (float*)alloc((size_t)nE * 4);
    float*  exb      = (float*)alloc((size_t)nE * 4);
    unsigned* maxkey = (unsigned*)alloc((size_t)NPn * 4);
    float*  ssum     = (float*)alloc((size_t)NPn * 4);
    float*  agg_p    = (float*)alloc((size_t)NPn * HID * 4);
    float*  agg_d    = (float*)alloc((size_t)ND * HID * 4);
    __bf16* gelu_d   = (__bf16*)alloc((size_t)ND * HID * 2);
    __bf16* gelu_p   = (__bf16*)alloc((size_t)NPn * HID * 2);
    float*  hd_f     = (float*)alloc((size_t)ND * HID * 4);
    float*  hp_f     = (float*)alloc((size_t)NPn * HID * 4);
    __bf16* hd_bf    = (__bf16*)alloc((size_t)ND * HID * 2);
    __bf16* hp_bf    = (__bf16*)alloc((size_t)NPn * HID * 2);

    auto blks = [](long n, int t) { return (int)((n + t - 1) / t); };

    // ---- 1. bf16 casts of A-side operands + transposed weight prep ----
    cast_bf16_kernel<<<blks((long)ND * FD, 256), 256, 0, stream>>>(x_drug, xdrug_bf, (long)ND * FD);
    cast_bf16_kernel<<<blks((long)NPn * FP, 256), 256, 0, stream>>>(x_prot, xprot_bf, (long)NPn * FP);
    cast_bf16_kernel<<<blks(2 * 128 * 128, 256), 256, 0, stream>>>(Wk, Wk_bf, 2 * 128 * 128);
    cast_bf16_kernel<<<blks(2 * 128 * 128, 256), 256, 0, stream>>>(Wv, Wv_bf, 2 * 128 * 128);
    transpose_cast_bf16_kernel<<<blks((long)128 * FD, 256), 256, 0, stream>>>(Wi_d, WiT_d, FD, 128, 128);
    transpose_cast_bf16_kernel<<<blks((long)128 * FP, 256), 256, 0, stream>>>(Wi_p, WiT_p, FP, 128, 128);
    for (int r_ = 0; r_ < 2; ++r_) {
        transpose_cast_bf16_kernel<<<blks(128 * 128, 256), 256, 0, stream>>>(
            Wq + r_ * 16384, WqT + r_ * 16384, 128, 128, 128);
        transpose_cast_bf16_kernel<<<blks(128 * 128, 256), 256, 0, stream>>>(
            Wo + r_ * 16384, WoT + r_ * 16384, 128, 128, 128);
        transpose_cast_bf16_kernel<<<blks(128 * 128, 256), 256, 0, stream>>>(
            a_rel + r_ * 16384, aT + r_ * 16384, 128, 128, 128);
        transpose_cast_bf16_kernel<<<blks(128 * 128, 256), 256, 0, stream>>>(
            m_rel + r_ * 16384, mT + r_ * 16384, 128, 128, 128);
    }
    transpose_cast_bf16_kernel<<<blks(16 * 256, 256), 256, 0, stream>>>(Wl, WlT, 256, OUT, 16);

    // ---- 2. folded biases + folded relation matrices (Wka = Wk@a, Wvm = Wv@m) ----
    for (int r_ = 0; r_ < 2; ++r_) {
        vecmat128_kernel<<<1, 128, 0, stream>>>(bk + r_ * 128, a_rel + r_ * 16384, bka + r_ * 128);
        vecmat128_kernel<<<1, 128, 0, stream>>>(bv + r_ * 128, m_rel + r_ * 16384, bvm + r_ * 128);
        gemm_bf16_wmma<<<dim3(1, 1), 256, 0, stream>>>(
            Wk_bf + r_ * 16384, aT + r_ * 16384, nullptr, nullptr, nullptr,
            Wka + r_ * 16384, nullptr, 128, 128, 128, 128);
        gemm_bf16_wmma<<<dim3(1, 1), 256, 0, stream>>>(
            Wv_bf + r_ * 16384, mT + r_ * 16384, nullptr, nullptr, nullptr,
            Wvm + r_ * 16384, nullptr, 128, 128, 128, 128);
        transpose_cast_bf16_kernel<<<blks(128 * 128, 256), 256, 0, stream>>>(
            Wka + r_ * 16384, WkaT + r_ * 16384, 128, 128, 128);
        transpose_cast_bf16_kernel<<<blks(128 * 128, 256), 256, 0, stream>>>(
            Wvm + r_ * 16384, WvmT + r_ * 16384, 128, 128, 128);
    }

    // ---- 3. input projections ----
    dim3 gnode(1, (ND + BM - 1) / BM);
    gemm_bf16_wmma<<<gnode, 256, 0, stream>>>(xdrug_bf, WiT_d, bi_d, nullptr, nullptr,
                                              xd_f, xd_bf, ND, HID, HID, FD);
    gemm_bf16_wmma<<<gnode, 256, 0, stream>>>(xprot_bf, WiT_p, bi_p, nullptr, nullptr,
                                              xp_f, xp_bf, NPn, HID, HID, FP);

    // ---- 4. Q + folded K/V per node type ----
    gemm_bf16_wmma<<<gnode, 256, 0, stream>>>(xd_bf, WqT,          bq,        nullptr, nullptr,
                                              q_d,  nullptr, ND, HID, HID, HID);
    gemm_bf16_wmma<<<gnode, 256, 0, stream>>>(xp_bf, WqT + 16384,  bq + 128,  nullptr, nullptr,
                                              q_p,  nullptr, NPn, HID, HID, HID);
    gemm_bf16_wmma<<<gnode, 256, 0, stream>>>(xd_bf, WkaT,         bka,       nullptr, nullptr,
                                              kp_d, nullptr, ND, HID, HID, HID);
    gemm_bf16_wmma<<<gnode, 256, 0, stream>>>(xd_bf, WvmT,         bvm,       nullptr, nullptr,
                                              vp_d, nullptr, ND, HID, HID, HID);
    gemm_bf16_wmma<<<gnode, 256, 0, stream>>>(xp_bf, WkaT + 16384, bka + 128, nullptr, nullptr,
                                              kp_p, nullptr, NPn, HID, HID, HID);
    gemm_bf16_wmma<<<gnode, 256, 0, stream>>>(xp_bf, WvmT + 16384, bvm + 128, nullptr, nullptr,
                                              vp_p, nullptr, NPn, HID, HID, HID);

    // ---- 5. edge phase ----
    const int ewaves = blks((long)nE * 32, 256);
    // relation 0: drug -> protein  (src=e_dp[0], dst=e_dp[1])
    fill_u32_kernel<<<blks(NPn, 256), 256, 0, stream>>>(maxkey, NEG_INF_KEY, NPn);
    fill_f32_kernel<<<blks(NPn, 256), 256, 0, stream>>>(ssum, 0.f, NPn);
    fill_f32_kernel<<<blks((long)NPn * HID, 256), 256, 0, stream>>>(agg_p, 0.f, (long)NPn * HID);
    edge_alpha_max_kernel<<<ewaves, 256, 0, stream>>>(q_p, kp_d, e_dp, e_dp + nE, p_rel,
                                                      alpha, maxkey, nE);
    edge_exp_sum_kernel<<<blks(nE, 256), 256, 0, stream>>>(alpha, e_dp + nE, maxkey, exb, ssum, nE);
    edge_scatter_kernel<<<ewaves, 256, 0, stream>>>(vp_d, e_dp, e_dp + nE, exb, ssum, agg_p, nE);
    // relation 1: protein -> drug
    fill_u32_kernel<<<blks(ND, 256), 256, 0, stream>>>(maxkey, NEG_INF_KEY, ND);
    fill_f32_kernel<<<blks(ND, 256), 256, 0, stream>>>(ssum, 0.f, ND);
    fill_f32_kernel<<<blks((long)ND * HID, 256), 256, 0, stream>>>(agg_d, 0.f, (long)ND * HID);
    edge_alpha_max_kernel<<<ewaves, 256, 0, stream>>>(q_d, kp_p, e_pd, e_pd + nE, p_rel + 1,
                                                      alpha, maxkey, nE);
    edge_exp_sum_kernel<<<blks(nE, 256), 256, 0, stream>>>(alpha, e_pd + nE, maxkey, exb, ssum, nE);
    edge_scatter_kernel<<<ewaves, 256, 0, stream>>>(vp_p, e_pd, e_pd + nE, exb, ssum, agg_d, nE);

    // ---- 6. gelu + node update (gated residual) ----
    gelu_bf16_kernel<<<blks((long)ND * HID, 256), 256, 0, stream>>>(agg_d, gelu_d, (long)ND * HID);
    gelu_bf16_kernel<<<blks((long)NPn * HID, 256), 256, 0, stream>>>(agg_p, gelu_p, (long)NPn * HID);
    gemm_bf16_wmma<<<gnode, 256, 0, stream>>>(gelu_d, WoT,         bo,       xd_f, skip,
                                              hd_f, hd_bf, ND, HID, HID, HID);
    gemm_bf16_wmma<<<gnode, 256, 0, stream>>>(gelu_p, WoT + 16384, bo + 128, xp_f, skip + 1,
                                              hp_f, hp_bf, NPn, HID, HID, HID);

    // ---- 7. link predictor: gather concat(hd[e0], hp[e1]) @ Wl + bl ----
    dim3 glink(1, (EL + LBM - 1) / LBM);
    link_gemm_wmma<<<glink, 256, 0, stream>>>(hd_bf, hp_bf, e_lbl, e_lbl + EL,
                                              WlT, bl, out, EL, OUT, 2 * HID);
}
